// RingDilatedAttentionHilbertCore_69776038690880
// MI455X (gfx1250) — matrix operand order repaired and checked by reading, not compiled, not run on hardware
//
#include <hip/hip_runtime.h>
#include <hip/hip_bf16.h>

// ---------------------------------------------------------------------------
// Ring dilated attention w/ Hilbert permutation, CDNA5 (gfx1250) WMMA version.
//   x = q.reshape(S, 512); 3 segments (2048/4096/8192 tokens, dilation 1/2/4)
//   per segment: hilbert-permute -> QKV gemm -> 7x(8 heads) of full 2048x2048
//   attention -> out gemm -> scatter through perm.
// All matmuls via v_wmma_f32_16x16x32_f16 (f32 accumulate).
// Workspace: perm (int[14336]) | qkv f16 [14336,1536] | oh f16 [14336,512]
// ---------------------------------------------------------------------------

typedef __attribute__((ext_vector_type(16))) _Float16 v16h;
typedef __attribute__((ext_vector_type(8)))  _Float16 v8h;
typedef __attribute__((ext_vector_type(8)))  float    v8f;
typedef __attribute__((ext_vector_type(4)))  float    v4f;

#define WMMA16(a, b, c) \
  __builtin_amdgcn_wmma_f32_16x16x32_f16(false, (a), false, (b), (short)0, (c), false, false)

static __device__ __forceinline__ v16h pack16(v8h lo, v8h hi) {
  v16h o;
#pragma unroll
  for (int e = 0; e < 8; ++e) { o[e] = lo[e]; o[8 + e] = hi[e]; }
  return o;
}

// A-fragment (16x32 f16 tile slice, row = this lane's m):
//   K = kbase + 16*hi + e (contiguous)
static __device__ __forceinline__ v16h frag_a(const _Float16* rowptr, int kbase, int hi) {
  const _Float16* p = rowptr + kbase + hi * 16;
  return pack16(*(const v8h*)p, *(const v8h*)(p + 8));
}

// B-fragment (32x16 f16): lane holds column n; 16 contiguous K values at kbase
// (source stored n-major so the K run is contiguous).
static __device__ __forceinline__ v16h frag_b(const _Float16* colptr, int kbase) {
  const _Float16* p = colptr + kbase;
  return pack16(*(const v8h*)p, *(const v8h*)(p + 8));
}

static __device__ __forceinline__ void seg_info(int seg, int& pos, int& L, int& r) {
  if (seg == 0)      { pos = 0;    L = 2048; r = 1; }
  else if (seg == 1) { pos = 2048; L = 4096; r = 2; }
  else               { pos = 6144; L = 8192; r = 4; }
}

static __device__ __forceinline__ void decode_mtile(int mt, int& seg, int& mbase) {
  if (mt < 16)      { seg = 0; mbase = mt * 128; }
  else if (mt < 48) { seg = 1; mbase = (mt - 16) * 128; }
  else              { seg = 2; mbase = (mt - 48) * 128; }
}

// ---------------------------------------------------------------------------
// Hilbert curve: distance d -> cell index (y*side + x)
// ---------------------------------------------------------------------------
static __device__ __forceinline__ int hilbert_d2cell(int d, int side) {
  int t = d, x = 0, y = 0;
  for (int s = 1; s < side; s <<= 1) {
    int rx = 1 & (t >> 1);
    int ry = 1 & (t ^ rx);
    if (ry == 0) {
      if (rx == 1) { x = s - 1 - x; y = s - 1 - y; }
      int tmp = x; x = y; y = tmp;
    }
    x += s * rx;
    y += s * ry;
    t >>= 2;
  }
  return y * side + x;
}

// One block per segment: stream d over the 2^k x 2^k grid, keep cells < L,
// block-wide stream compaction via wave32 ballot + popc scan.
__global__ __launch_bounds__(256) void build_perm_kernel(int* __restrict__ perm) {
  int seg = blockIdx.x;
  int pos, L, r;
  seg_info(seg, pos, L, r);
  (void)r;
  int side = (seg == 2) ? 128 : 64;
  int* p = perm + pos;

  __shared__ int s_wsum[8];
  __shared__ int s_running;
  int t = threadIdx.x, wid = t >> 5, lane = t & 31;
  if (t == 0) s_running = 0;

  int total = side * side;
  for (int start = 0; start < total; start += 256) {
    __syncthreads();
    int d = start + t;
    int cell = 0, valid = 0;
    if (d < total) { cell = hilbert_d2cell(d, side); valid = (cell < L); }
    unsigned long long bal = __ballot(valid);
    int wprefix = __popcll(bal & ((1ull << lane) - 1ull));
    if (lane == 0) s_wsum[wid] = __popcll(bal);
    __syncthreads();
    int base = s_running;
    for (int w = 0; w < wid; ++w) base += s_wsum[w];
    if (valid) p[base + wprefix] = cell;
    __syncthreads();
    if (t == 0) {
      int tot = 0;
      for (int w = 0; w < 8; ++w) tot += s_wsum[w];
      s_running += tot;
    }
  }
}

// ---------------------------------------------------------------------------
// QKV projection: qkv[pos+j, :] = x[pos+perm[j], :] @ Wqkv[seg] + bqkv[seg]
// Block tile 128x64, BK=32, 8 waves -> each wave a 32x32 tile (2x2 WMMA).
// ---------------------------------------------------------------------------
__global__ __launch_bounds__(256)
void qkv_gemm_kernel(const float* __restrict__ x, const float* __restrict__ W,
                     const float* __restrict__ bias, const int* __restrict__ perm,
                     _Float16* __restrict__ qkv) {
  __shared__ _Float16 As[128 * 40];   // A tile, row-major, pitch 40
  __shared__ _Float16 Bs[64 * 40];    // B tile, n-major (transposed), pitch 40
  __shared__ int rowIdx[128];

  int t = threadIdx.x;
  int seg, mbase;
  decode_mtile(blockIdx.x, seg, mbase);
  int pos, L, r;
  seg_info(seg, pos, L, r);
  (void)L; (void)r;
  int nbase = blockIdx.y * 64;
  const float* Wseg = W + (size_t)seg * 512 * 1536;
  const float* bseg = bias + seg * 1536;

  if (t < 128) rowIdx[t] = pos + perm[pos + mbase + t];

  int lane = t & 31, wid = t >> 5;
  int ln = lane & 15, hi = lane >> 4;
  int wm = (wid >> 1) * 32, wn = (wid & 1) * 32;

  v8f acc[2][2];
#pragma unroll
  for (int a = 0; a < 2; ++a)
#pragma unroll
    for (int b = 0; b < 2; ++b)
#pragma unroll
      for (int j = 0; j < 8; ++j) acc[a][b][j] = 0.0f;

  for (int kk = 0; kk < 512; kk += 32) {
    __syncthreads();
    {  // A: 128x32, gathered rows, f32 -> f16
      int rr = t >> 1, cb = (t & 1) * 16;
      const float* src = x + (size_t)rowIdx[rr] * 512 + kk + cb;
      v4f f0 = *(const v4f*)src,       f1 = *(const v4f*)(src + 4);
      v4f f2 = *(const v4f*)(src + 8), f3 = *(const v4f*)(src + 12);
      v8h h0, h1;
#pragma unroll
      for (int e = 0; e < 4; ++e) {
        h0[e] = (_Float16)f0[e]; h0[4 + e] = (_Float16)f1[e];
        h1[e] = (_Float16)f2[e]; h1[4 + e] = (_Float16)f3[e];
      }
      *(v8h*)&As[rr * 40 + cb] = h0;
      *(v8h*)&As[rr * 40 + cb + 8] = h1;
    }
    {  // B: 32x64 -> Bs[n][k] transposed, f32 -> f16
      int k = t >> 3, nb = (t & 7) * 8;
      const float* src = Wseg + (size_t)(kk + k) * 1536 + nbase + nb;
      v4f f0 = *(const v4f*)src, f1 = *(const v4f*)(src + 4);
#pragma unroll
      for (int e = 0; e < 4; ++e) {
        Bs[(nb + e) * 40 + k]     = (_Float16)f0[e];
        Bs[(nb + 4 + e) * 40 + k] = (_Float16)f1[e];
      }
    }
    __syncthreads();
    v16h a0 = frag_a(&As[(wm + ln) * 40], 0, hi);
    v16h a1 = frag_a(&As[(wm + 16 + ln) * 40], 0, hi);
    v16h b0 = frag_b(&Bs[(wn + ln) * 40], hi * 16);
    v16h b1 = frag_b(&Bs[(wn + 16 + ln) * 40], hi * 16);
    acc[0][0] = WMMA16(a0, b0, acc[0][0]);
    acc[0][1] = WMMA16(a0, b1, acc[0][1]);
    acc[1][0] = WMMA16(a1, b0, acc[1][0]);
    acc[1][1] = WMMA16(a1, b1, acc[1][1]);
  }

#pragma unroll
  for (int ms = 0; ms < 2; ++ms)
#pragma unroll
    for (int ns = 0; ns < 2; ++ns) {
      int col = nbase + wn + ns * 16 + ln;
      float bv = bseg[col];
#pragma unroll
      for (int j = 0; j < 8; ++j) {
        int m = mbase + wm + ms * 16 + j + 8 * hi;
        qkv[(size_t)(pos + m) * 1536 + col] = (_Float16)(acc[ms][ns][j] + bv);
      }
    }
}

// ---------------------------------------------------------------------------
// Flash attention per (segment, dilation group, head); group length = 2048.
// Block = 8 waves x 16 query rows = 128 queries; key tiles of 32.
// ---------------------------------------------------------------------------
__global__ __launch_bounds__(256)
void attn_kernel(const _Float16* __restrict__ qkv, _Float16* __restrict__ oh) {
  __shared__ _Float16 Ks[32 * 72];      // K tile [token][hd], pitch 72
  __shared__ _Float16 Vt[64 * 40];      // V tile transposed [hd][token], pitch 40
  __shared__ _Float16 Ps[8][16 * 40];   // per-wave P bounce, pitch 40

  int z = blockIdx.z;
  int seg, g;
  if (z < 1)      { seg = 0; g = 0; }
  else if (z < 3) { seg = 1; g = z - 1; }
  else            { seg = 2; g = z - 3; }
  int pos, L, r;
  seg_info(seg, pos, L, r);
  (void)L;
  int h = blockIdx.y;
  int t = threadIdx.x, lane = t & 31, wid = t >> 5;
  int ln = lane & 15, hi = lane >> 4;
  int q0 = blockIdx.x * 128 + wid * 16;

  // Q fragments straight from global (row-major qkv, contiguous K runs)
  const _Float16* pq = qkv + (size_t)(pos + (size_t)(q0 + ln) * r + g) * 1536 + h * 64;
  v16h qf0 = frag_a(pq, 0, hi);
  v16h qf1 = frag_a(pq, 32, hi);

  v8f O[4];
  float mrow[8], lrow[8];
#pragma unroll
  for (int nd = 0; nd < 4; ++nd)
#pragma unroll
    for (int j = 0; j < 8; ++j) O[nd][j] = 0.0f;
#pragma unroll
  for (int j = 0; j < 8; ++j) { mrow[j] = -3.0e38f; lrow[j] = 0.0f; }

  const float scale = 0.125f;  // 1/sqrt(64)
  _Float16* PsW = &Ps[wid][0];

  for (int kt = 0; kt < 64; ++kt) {
    __syncthreads();
    {  // stage K (row-major) and V (transposed) tiles
      int tt = t >> 3, cb = (t & 7) * 8;
      size_t row = (size_t)(pos + (size_t)(kt * 32 + tt) * r + g) * 1536 + h * 64 + cb;
      v8h kv = *(const v8h*)(qkv + row + 512);
      *(v8h*)&Ks[tt * 72 + cb] = kv;
      v8h vv = *(const v8h*)(qkv + row + 1024);
#pragma unroll
      for (int e = 0; e < 8; ++e) Vt[(cb + e) * 40 + tt] = vv[e];
    }
    __syncthreads();
    if (kt + 1 < 64) {  // prefetch next K tile row (global_prefetch_b8)
      int tt = t >> 3, cb = (t & 7) * 8;
      __builtin_prefetch(
          qkv + (size_t)(pos + (size_t)((kt + 1) * 32 + tt) * r + g) * 1536 + h * 64 + 512 + cb,
          0, 0);
    }

    // scores S[16q x 32k] = Q[16x64] @ K^T : two hd chunks per key subtile
    v8f S0, S1;
#pragma unroll
    for (int j = 0; j < 8; ++j) { S0[j] = 0.0f; S1[j] = 0.0f; }
    v16h kb00 = frag_b(&Ks[ln * 72],        0  + hi * 16);
    v16h kb01 = frag_b(&Ks[ln * 72],        32 + hi * 16);
    v16h kb10 = frag_b(&Ks[(16 + ln) * 72], 0  + hi * 16);
    v16h kb11 = frag_b(&Ks[(16 + ln) * 72], 32 + hi * 16);
    S0 = WMMA16(qf0, kb00, S0);
    S0 = WMMA16(qf1, kb01, S0);
    S1 = WMMA16(qf0, kb10, S1);
    S1 = WMMA16(qf1, kb11, S1);

    // online softmax; row m = j + 8*hi lives in this 16-lane half
#pragma unroll
    for (int j = 0; j < 8; ++j) {
      float s0 = S0[j] * scale, s1 = S1[j] * scale;
      float mx = fmaxf(s0, s1);
#pragma unroll
      for (int off = 8; off >= 1; off >>= 1) mx = fmaxf(mx, __shfl_xor(mx, off, 32));
      float mnew = fmaxf(mrow[j], mx);
      float alpha = __expf(mrow[j] - mnew);
      mrow[j] = mnew;
      float p0 = __expf(s0 - mnew), p1 = __expf(s1 - mnew);
      float rs = p0 + p1;
#pragma unroll
      for (int off = 8; off >= 1; off >>= 1) rs += __shfl_xor(rs, off, 32);
      lrow[j] = lrow[j] * alpha + rs;
#pragma unroll
      for (int nd = 0; nd < 4; ++nd) O[nd][j] *= alpha;
      int m = j + 8 * hi;
      PsW[m * 40 + ln]      = (_Float16)p0;
      PsW[m * 40 + 16 + ln] = (_Float16)p1;
    }
    // per-wave LDS bounce: LDS is in-order per wave; wait for stores to land
    asm volatile("s_wait_dscnt 0x0" ::: "memory");
    v16h pa = frag_a(&PsW[ln * 40], 0, hi);
#pragma unroll
    for (int nd = 0; nd < 4; ++nd) {
      v16h vb = frag_b(&Vt[(nd * 16 + ln) * 40], hi * 16);
      O[nd] = WMMA16(pa, vb, O[nd]);
    }
  }

  // normalize + store f16 attention output, permuted-token row = i*r + g
#pragma unroll
  for (int j = 0; j < 8; ++j) {
    float invl = 1.0f / lrow[j];
    int m = q0 + j + 8 * hi;
    size_t row = (size_t)(pos + (size_t)m * r + g) * 512 + h * 64;
#pragma unroll
    for (int nd = 0; nd < 4; ++nd)
      oh[row + nd * 16 + ln] = (_Float16)(O[nd][j] * invl);
  }
}

// ---------------------------------------------------------------------------
// Output projection + scatter: out[pos+perm[j], :] = oh[pos+j, :] @ Wout + bout
// ---------------------------------------------------------------------------
__global__ __launch_bounds__(256)
void out_gemm_kernel(const _Float16* __restrict__ oh, const float* __restrict__ W,
                     const float* __restrict__ bias, const int* __restrict__ perm,
                     float* __restrict__ out) {
  __shared__ _Float16 As[128 * 40];
  __shared__ _Float16 Bs[64 * 40];
  __shared__ int outRow[128];

  int t = threadIdx.x;
  int seg, mbase;
  decode_mtile(blockIdx.x, seg, mbase);
  int pos, L, r;
  seg_info(seg, pos, L, r);
  (void)L; (void)r;
  int nbase = blockIdx.y * 64;
  const float* Wseg = W + (size_t)seg * 512 * 512;
  const float* bseg = bias + seg * 512;

  if (t < 128) outRow[t] = pos + perm[pos + mbase + t];

  int lane = t & 31, wid = t >> 5;
  int ln = lane & 15, hi = lane >> 4;
  int wm = (wid >> 1) * 32, wn = (wid & 1) * 32;

  v8f acc[2][2];
#pragma unroll
  for (int a = 0; a < 2; ++a)
#pragma unroll
    for (int b = 0; b < 2; ++b)
#pragma unroll
      for (int j = 0; j < 8; ++j) acc[a][b][j] = 0.0f;

  for (int kk = 0; kk < 512; kk += 32) {
    __syncthreads();
    {  // A: f16 already, straight copy
      int rr = t >> 1, cb = (t & 1) * 16;
      const _Float16* src = oh + (size_t)(pos + mbase + rr) * 512 + kk + cb;
      *(v8h*)&As[rr * 40 + cb]     = *(const v8h*)src;
      *(v8h*)&As[rr * 40 + cb + 8] = *(const v8h*)(src + 8);
    }
    {  // B: Wout f32 -> f16, transposed into Bs[n][k]
      int k = t >> 3, nb = (t & 7) * 8;
      const float* src = Wseg + (size_t)(kk + k) * 512 + nbase + nb;
      v4f f0 = *(const v4f*)src, f1 = *(const v4f*)(src + 4);
#pragma unroll
      for (int e = 0; e < 4; ++e) {
        Bs[(nb + e) * 40 + k]     = (_Float16)f0[e];
        Bs[(nb + 4 + e) * 40 + k] = (_Float16)f1[e];
      }
    }
    __syncthreads();
    v16h a0 = frag_a(&As[(wm + ln) * 40], 0, hi);
    v16h a1 = frag_a(&As[(wm + 16 + ln) * 40], 0, hi);
    v16h b0 = frag_b(&Bs[(wn + ln) * 40], hi * 16);
    v16h b1 = frag_b(&Bs[(wn + 16 + ln) * 40], hi * 16);
    acc[0][0] = WMMA16(a0, b0, acc[0][0]);
    acc[0][1] = WMMA16(a0, b1, acc[0][1]);
    acc[1][0] = WMMA16(a1, b0, acc[1][0]);
    acc[1][1] = WMMA16(a1, b1, acc[1][1]);
  }

#pragma unroll
  for (int ms = 0; ms < 2; ++ms)
#pragma unroll
    for (int ns = 0; ns < 2; ++ns) {
      int col = nbase + wn + ns * 16 + ln;
      float bv = bseg[col];
#pragma unroll
      for (int j = 0; j < 8; ++j) {
        int m = wm + ms * 16 + j + 8 * hi;  // local row in tile
        out[(size_t)outRow[m] * 512 + col] = acc[ms][ns][j] + bv;
      }
    }
}

// ---------------------------------------------------------------------------
extern "C" void kernel_launch(void* const* d_in, const int* in_sizes, int n_in,
                              void* d_out, int out_size, void* d_ws, size_t ws_size,
                              hipStream_t stream) {
  (void)in_sizes; (void)n_in; (void)out_size; (void)ws_size;
  const float* x    = (const float*)d_in[0];  // q -> activation; k,v unused
  const float* Wqkv = (const float*)d_in[3];
  const float* bqkv = (const float*)d_in[4];
  const float* Wout = (const float*)d_in[5];
  const float* bout = (const float*)d_in[6];

  // workspace layout (needs ~57 MB):
  // [0, 57344)            perm  int32[14336]
  // [64KiB, +44040192)    qkv   f16 [14336,1536]
  // [.., +14680064)       oh    f16 [14336,512]
  int*      perm = (int*)d_ws;
  _Float16* qkv  = (_Float16*)((char*)d_ws + (1 << 16));
  _Float16* oh   = (_Float16*)((char*)d_ws + (1 << 16) + (size_t)14336 * 1536 * 2);

  build_perm_kernel<<<3, 256, 0, stream>>>(perm);
  qkv_gemm_kernel<<<dim3(112, 24), 256, 0, stream>>>(x, Wqkv, bqkv, perm, qkv);
  attn_kernel<<<dim3(16, 8, 7), 256, 0, stream>>>(qkv, oh);
  out_gemm_kernel<<<dim3(112, 8), 256, 0, stream>>>(oh, Wout, bout, perm, (float*)d_out);
}
